// SimDistillLoss_36593121362411
// MI455X (gfx1250) — compile-verified, matrix operand order checked by reading, not run on hardware
//
#include <hip/hip_runtime.h>

// Problem constants (from reference): view [N, V, D] f32, simY [N, N] f32.
#define NROWS 8192
#define NVIEW 2
#define DDIM  128
#define TILE  128          // sims tile edge per workgroup
#define LDSB_STRIDE 136    // padded f16 row stride in LDS (272B -> bank spread)

typedef _Float16 f16;
typedef __attribute__((ext_vector_type(16))) _Float16 v16h;
typedef __attribute__((ext_vector_type(8)))  _Float16 v8h;
typedef __attribute__((ext_vector_type(4)))  _Float16 v4h;
typedef __attribute__((ext_vector_type(8)))  float    v8f;
typedef __attribute__((ext_vector_type(4)))  float    v4f;

__device__ __forceinline__ float wave_reduce_add(float v) {
#pragma unroll
    for (int off = 16; off > 0; off >>= 1)
        v += __shfl_xor(v, off, 32);
    return v;
}

// ---------------------------------------------------------------------------
// Kernel 1: row-normalize view[n][v][:] (D=128) and store f16 as xn[v][n][:].
// One wave32 per (n, v) row; each lane handles 4 contiguous floats.
// ---------------------------------------------------------------------------
__global__ void __launch_bounds__(256)
normalize_f16_kernel(const float* __restrict__ view, f16* __restrict__ xn) {
    const int wave = (blockIdx.x * blockDim.x + threadIdx.x) >> 5;
    const int lane = threadIdx.x & 31;
    const int v = wave & 1;
    const int n = wave >> 1;
    if (n >= NROWS) return;

    const float* src = view + ((size_t)n * NVIEW + v) * DDIM + lane * 4;
    v4f x = *(const v4f*)src;
    float ss = x[0] * x[0] + x[1] * x[1] + x[2] * x[2] + x[3] * x[3];
    ss = wave_reduce_add(ss);
    const float inv = 1.0f / fmaxf(sqrtf(ss), 1e-8f);

    v4h o;
    o[0] = (f16)(x[0] * inv);
    o[1] = (f16)(x[1] * inv);
    o[2] = (f16)(x[2] * inv);
    o[3] = (f16)(x[3] * inv);
    f16* dst = xn + ((size_t)v * NROWS + n) * DDIM + lane * 4;
    *(v4h*)dst = o;
}

// ---------------------------------------------------------------------------
// Kernel 2: fused dual-view Gram tile + squared-error vs simY.
// Block = 256 threads (8 waves), tile = 128x128 of sims, both views at once
// so each simY element is read exactly once from HBM.
// Wave w computes rows [16w, 16w+16) of the tile: 8 col subtiles x 4 K-steps
// x 2 views = 64 v_wmma_f32_16x16x32_f16 per wave.
// ---------------------------------------------------------------------------
__global__ void __launch_bounds__(256)
simloss_wmma_kernel(const f16* __restrict__ xn, const float* __restrict__ simY,
                    float* __restrict__ partials) {
    extern __shared__ f16 ldsB[];  // [NVIEW][TILE][LDSB_STRIDE] = 69632 B

    const int tid  = threadIdx.x;
    const int lane = tid & 31;
    const int wave = tid >> 5;
    const int col0 = blockIdx.x * TILE;   // tile column origin (B rows)
    const int row0 = blockIdx.y * TILE;   // tile row origin    (A rows)

    // Cooperative stage of the column strip (both views) into LDS.
    {
        const int v = tid >> 7;      // 0..1
        const int r = tid & 127;     // row of strip
        const f16* src = xn + ((size_t)v * NROWS + col0 + r) * DDIM;
        f16* dst = ldsB + ((size_t)v * TILE + r) * LDSB_STRIDE;
#pragma unroll
        for (int i = 0; i < DDIM / 8; ++i)
            *(v8h*)(dst + i * 8) = *(const v8h*)(src + i * 8);
    }
    __syncthreads();

    const int lh  = lane & 15;   // row (A) / col (B) within fragment
    const int sel = lane >> 4;   // K-half select per 16-bit WMMA layout
    const int m0  = row0 + wave * 16;

    v8f acc0[8] = {};
    v8f acc1[8] = {};

    const f16* aRow0 = xn + (size_t)(m0 + lh) * DDIM;                        // view 0
    const f16* aRow1 = xn + ((size_t)NROWS + (size_t)(m0 + lh)) * DDIM;      // view 1

#pragma unroll
    for (int k0 = 0; k0 < DDIM; k0 += 32) {
        // A fragments (16x32 f16): lanes<16 take K {0..7,16..23}, lanes>=16 {8..15,24..31}
        v16h a0, a1;
        {
            v8h c0 = *(const v8h*)(aRow0 + k0 + sel * 8);
            v8h c1 = *(const v8h*)(aRow0 + k0 + sel * 8 + 16);
#pragma unroll
            for (int i = 0; i < 8; ++i) { a0[i] = c0[i]; a0[8 + i] = c1[i]; }
        }
        {
            v8h c0 = *(const v8h*)(aRow1 + k0 + sel * 8);
            v8h c1 = *(const v8h*)(aRow1 + k0 + sel * 8 + 16);
#pragma unroll
            for (int i = 0; i < 8; ++i) { a1[i] = c0[i]; a1[8 + i] = c1[i]; }
        }
#pragma unroll
        for (int c = 0; c < 8; ++c) {
            // B fragment (32x16): B[k][n] = Xn[col0 + n][k]; same K packing as A.
            const f16* bp0 = ldsB + (size_t)(c * 16 + lh) * LDSB_STRIDE + k0 + sel * 8;
            const f16* bp1 = bp0 + (size_t)TILE * LDSB_STRIDE;
            v16h b0, b1;
            {
                v8h lo = *(const v8h*)bp0;
                v8h hi = *(const v8h*)(bp0 + 16);
#pragma unroll
                for (int i = 0; i < 8; ++i) { b0[i] = lo[i]; b0[8 + i] = hi[i]; }
            }
            {
                v8h lo = *(const v8h*)bp1;
                v8h hi = *(const v8h*)(bp1 + 16);
#pragma unroll
                for (int i = 0; i < 8; ++i) { b1[i] = lo[i]; b1[8 + i] = hi[i]; }
            }
            acc0[c] = __builtin_amdgcn_wmma_f32_16x16x32_f16(
                false, a0, false, b0, (short)0, acc0[c], false, false);
            acc1[c] = __builtin_amdgcn_wmma_f32_16x16x32_f16(
                false, a1, false, b1, (short)0, acc1[c], false, false);
        }
    }

    // Fused loss: C layout -> VGPR r, lane: (M = r + 8*sel, N = lh).
    float lsum = 0.0f;
#pragma unroll
    for (int c = 0; c < 8; ++c) {
        const float* y = simY + (size_t)(m0 + sel * 8) * NROWS + col0 + c * 16 + lh;
#pragma unroll
        for (int r = 0; r < 8; ++r) {
            const float yv = y[(size_t)r * NROWS];
            const float e0 = acc0[c][r] - yv;
            const float e1 = acc1[c][r] - yv;
            lsum += e0 * e0 + e1 * e1;
        }
    }

    lsum = wave_reduce_add(lsum);
    __syncthreads();                 // LDS reads done; safe to reuse for reduction
    float* red = (float*)ldsB;
    if (lane == 0) red[wave] = lsum;
    __syncthreads();
    if (tid == 0) {
        float t = 0.0f;
#pragma unroll
        for (int w = 0; w < 8; ++w) t += red[w];
        partials[(size_t)blockIdx.y * gridDim.x + blockIdx.x] = t;
    }
}

// ---------------------------------------------------------------------------
// Kernel 3: deterministic final reduction of per-block partials -> scalar loss.
// loss = sum_v mean_{n,m} (sims - simY)^2  ==  (total sum) / N^2
// ---------------------------------------------------------------------------
__global__ void __launch_bounds__(256)
final_reduce_kernel(const float* __restrict__ partials, float* __restrict__ out, int n) {
    __shared__ float red[8];
    float s = 0.0f;
    for (int i = threadIdx.x; i < n; i += blockDim.x) s += partials[i];
    s = wave_reduce_add(s);
    const int lane = threadIdx.x & 31;
    const int wave = threadIdx.x >> 5;
    if (lane == 0) red[wave] = s;
    __syncthreads();
    if (threadIdx.x == 0) {
        float t = 0.0f;
#pragma unroll
        for (int w = 0; w < 8; ++w) t += red[w];
        out[0] = t / (float)((long long)NROWS * (long long)NROWS);
    }
}

extern "C" void kernel_launch(void* const* d_in, const int* in_sizes, int n_in,
                              void* d_out, int out_size, void* d_ws, size_t ws_size,
                              hipStream_t stream) {
    (void)in_sizes; (void)n_in; (void)out_size; (void)ws_size;
    const float* view = (const float*)d_in[0];   // [N, V, D] f32
    const float* simY = (const float*)d_in[1];   // [N, N]    f32
    float* out = (float*)d_out;

    f16* xn = (f16*)d_ws;                                              // 4 MB f16
    float* partials = (float*)((char*)d_ws +
                               (size_t)NVIEW * NROWS * DDIM * sizeof(f16));

    // Phase 1: normalize + f32 -> f16 (one wave per row; 16384 waves).
    {
        const int waves = NROWS * NVIEW;
        dim3 grid(waves / 8), block(256);
        normalize_f16_kernel<<<grid, block, 0, stream>>>(view, xn);
    }
    // Phase 2: fused dual-view WMMA Gram + loss tiles.
    {
        dim3 grid(NROWS / TILE, NROWS / TILE), block(256);
        const size_t shmem = (size_t)NVIEW * TILE * LDSB_STRIDE * sizeof(f16);
        simloss_wmma_kernel<<<grid, block, shmem, stream>>>(xn, simY, partials);
    }
    // Phase 3: final reduction to scalar.
    {
        const int nPart = (NROWS / TILE) * (NROWS / TILE);
        final_reduce_kernel<<<1, 256, 0, stream>>>(partials, out, nPart);
    }
}
